// TcLstm_79697413144840
// MI455X (gfx1250) — compile-verified, matrix operand order checked by reading, not run on hardware
//
#include <hip/hip_runtime.h>
#include <math.h>

#define Vv 50000
#define Ee 300
#define Ss 128
#define TTt 12
#define Hh 512
#define Cc 3
#define Bb 128

#define KPAD 608      // 600 padded to multiple of 32
#define N4H  2048     // 4*H
#define KH   512      // H
#define MROWS (Ss*Bb) // 16384
#define KC   32       // k-chunk staged in LDS per iteration

typedef __attribute__((ext_vector_type(16))) __bf16 v16bf;
typedef __attribute__((ext_vector_type(8)))  float  v8f;
typedef __attribute__((ext_vector_type(4)))  int    v4i;

// address-space-qualified element typedefs (attribute must live in a typedef,
// not in a cast type-name, or clang silently drops it)
typedef __attribute__((address_space(1))) v4i v4i_g;   // global
typedef __attribute__((address_space(3))) v4i v4i_l;   // LDS

union FragU { unsigned int u[8]; v16bf v; };

// ---- async global->LDS copy (CDNA5 GLOBAL_LOAD_ASYNC_TO_LDS_B128), with fallback ----
#if defined(__has_builtin)
#if __has_builtin(__builtin_amdgcn_global_load_async_to_lds_b128)
#define HAVE_ASYNC_LDS 1
#endif
#if __has_builtin(__builtin_amdgcn_s_wait_asynccnt)
#define HAVE_WAIT_ASYNC 1
#endif
#endif

#ifdef HAVE_ASYNC_LDS
#define ASYNC_B128(gsrc, ldst)                                                   \
    __builtin_amdgcn_global_load_async_to_lds_b128((v4i_g*)(gsrc), (v4i_l*)(ldst), 0, 0)
#ifdef HAVE_WAIT_ASYNC
#define ASYNC_WAIT() __builtin_amdgcn_s_wait_asynccnt(0)
#else
#define ASYNC_WAIT() asm volatile("s_wait_asynccnt 0x0" ::: "memory")
#endif
#else
#define ASYNC_B128(gsrc, ldst)                                                   \
    do { *(v4i*)(ldst) = *(const v4i*)(gsrc); } while (0)
#define ASYNC_WAIT() do {} while (0)
#endif

__device__ __forceinline__ unsigned short f2bf(float f) {
    unsigned u = __builtin_bit_cast(unsigned, f);
    unsigned r = u + 0x7FFFu + ((u >> 16) & 1u);   // round-to-nearest-even
    return (unsigned short)(r >> 16);
}

__device__ __forceinline__ float sigm(float x) { return 1.0f / (1.0f + __expf(-x)); }

// Load a 16x32 bf16 A/B fragment (ISA 7.12.2 16-bit A layout; B^T uses the same
// layout with N in place of M). p is row-major [rows][ld] ushort (bf16 bits).
// Works for both global and LDS (generic) pointers.
__device__ __forceinline__ v16bf load_frag(const unsigned short* p, int row0, int ld, int k0) {
    int lane = threadIdx.x & 31;
    int lr = lane & 15, hi = lane >> 4;
    const unsigned short* rp = p + (size_t)(row0 + lr) * ld + k0 + hi * 8;
    FragU f;
#pragma unroll
    for (int i = 0; i < 4; ++i) {
        f.u[i]     = *(const unsigned int*)(rp + 2 * i);        // K = 0..7  (+8*hi)
        f.u[i + 4] = *(const unsigned int*)(rp + 16 + 2 * i);   // K = 16..23(+8*hi)
    }
    return f.v;
}

// C/D f32 16x16 layout: VGPR r -> M = r + 8*(lane/16); N = lane%16
__device__ __forceinline__ v8f load_c(const float* p, int row0, int col0, int ld) {
    int lane = threadIdx.x & 31;
    int lr = lane & 15, hi = lane >> 4;
    v8f c;
#pragma unroll
    for (int r = 0; r < 8; ++r)
        c[r] = p[(size_t)(row0 + r + 8 * hi) * ld + col0 + lr];
    return c;
}

__device__ __forceinline__ void store_c(float* p, int row0, int col0, int ld, v8f c) {
    int lane = threadIdx.x & 31;
    int lr = lane & 15, hi = lane >> 4;
#pragma unroll
    for (int r = 0; r < 8; ++r)
        p[(size_t)(row0 + r + 8 * hi) * ld + col0 + lr] = c[r];
}

__device__ __forceinline__ v8f wmma_bf16(v16bf a, v16bf b, v8f c) {
    return __builtin_amdgcn_wmma_f32_16x16x32_bf16(false, a, false, b, (short)0, c, false, false);
}

__device__ __forceinline__ v8f zero8() {
    v8f z = {0.f, 0.f, 0.f, 0.f, 0.f, 0.f, 0.f, 0.f};
    return z;
}

// ---------------- mean of target embeddings ----------------
__global__ void mean_target_kernel(const int* __restrict__ ids, const int* __restrict__ lens,
                                   const float* __restrict__ emb, float* __restrict__ mean_t) {
    int b = blockIdx.x;
    int len = lens[b];
    for (int e = threadIdx.x; e < Ee; e += blockDim.x) {
        float s = 0.f;
        for (int t = 0; t < TTt; ++t)
            if (t < len) s += emb[(size_t)ids[b * TTt + t] * Ee + e];
        mean_t[b * Ee + e] = s / (float)len;
    }
}

// ---------------- weight transpose + bf16 convert ----------------
// WxT[n][k] = W[k][n], k<600 (pad to 608 with 0);  W is [1112][2048] row-major
__global__ void transpose_wx_kernel(const float* __restrict__ W, unsigned short* __restrict__ WxT) {
    int total = N4H * KPAD;
    for (int idx = blockIdx.x * blockDim.x + threadIdx.x; idx < total; idx += gridDim.x * blockDim.x) {
        int n = idx / KPAD, k = idx % KPAD;
        float v = (k < 600) ? W[(size_t)k * N4H + n] : 0.f;
        WxT[idx] = f2bf(v);
    }
}
// WhT[n][k] = W[600+k][n], k<512
__global__ void transpose_wh_kernel(const float* __restrict__ W, unsigned short* __restrict__ WhT) {
    int total = N4H * KH;
    for (int idx = blockIdx.x * blockDim.x + threadIdx.x; idx < total; idx += gridDim.x * blockDim.x) {
        int n = idx / KH, k = idx % KH;
        WhT[idx] = f2bf(W[(size_t)(600 + k) * N4H + n]);
    }
}

// ---------------- build gathered bf16 input matrix ----------------
// X[m][k], m = s*B + b : [emb(ids[b][s]) | mean_t[b] | 0-pad]
__global__ void build_x_kernel(const int* __restrict__ ids, const float* __restrict__ emb,
                               const float* __restrict__ mean_t, unsigned short* __restrict__ X) {
    int m = blockIdx.x;
    int s = m >> 7;        // /Bb
    int b = m & (Bb - 1);
    int id = ids[b * Ss + s];
    for (int k = threadIdx.x; k < KPAD; k += blockDim.x) {
        float v;
        if (k < Ee)            v = emb[(size_t)id * Ee + k];
        else if (k < 2 * Ee)   v = mean_t[b * Ee + (k - Ee)];
        else                   v = 0.f;
        X[(size_t)m * KPAD + k] = f2bf(v);
    }
}

// ---------------- big GEMM: Zx = X @ WxT^T + bias  (M=16384,N=2048,K=608) ----------------
// block = 256 threads (8 waves); macro-tile 64 rows x 256 cols; per wave 32x64 (2x4 tiles).
// Panels staged in LDS per 32-wide k-chunk via async global->LDS b128 copies.
__global__ void gemm_zx_kernel(const unsigned short* __restrict__ X,
                               const unsigned short* __restrict__ WxT,
                               const float* __restrict__ bias,
                               float* __restrict__ Zx) {
    __shared__ unsigned short As[64 * KC];    //  4 KB : 64 rows  x 32 k
    __shared__ unsigned short Bs[256 * KC];   // 16 KB : 256 cols x 32 k

    int tid = threadIdx.x;
    int wid = tid >> 5;
    int rowBlock = blockIdx.x * 64;
    int colBlock = blockIdx.y * 256;
    int r0 = (wid & 1) * 32;      // wave row offset inside macro-tile
    int c0 = (wid >> 1) * 64;     // wave col offset inside macro-tile

    v8f acc[2][4];
#pragma unroll
    for (int i = 0; i < 2; ++i)
#pragma unroll
        for (int j = 0; j < 4; ++j) acc[i][j] = zero8();

    // per-thread copy assignment: chunk = 8 ushorts (16 B)
    int crow = tid >> 2;          // 0..63
    int cpart = tid & 3;          // 0..3  (which 16B of the 64B row-chunk)
    const unsigned short* agp = X + (size_t)(rowBlock + crow) * KPAD + cpart * 8;
    const unsigned short* bgp = WxT + (size_t)(colBlock + crow) * KPAD + cpart * 8;
    unsigned short* alp = &As[crow * KC + cpart * 8];
    unsigned short* blp = &Bs[crow * KC + cpart * 8];

    for (int kk = 0; kk < KPAD / KC; ++kk) {
        int k0 = kk * KC;
        // stage A panel: 64 rows x 32 k  = 256 b128 chunks (1 per thread)
        ASYNC_B128(agp + k0, alp);
        // stage B panel: 256 rows x 32 k = 1024 b128 chunks (4 per thread)
#pragma unroll
        for (int i = 0; i < 4; ++i)
            ASYNC_B128(bgp + (size_t)i * 64 * KPAD + k0, blp + i * 64 * KC);
        ASYNC_WAIT();
        __syncthreads();

        v16bf af0 = load_frag(As, r0, KC, 0);
        v16bf af1 = load_frag(As, r0 + 16, KC, 0);
#pragma unroll
        for (int j = 0; j < 4; ++j) {
            v16bf bf = load_frag(Bs, c0 + j * 16, KC, 0);
            acc[0][j] = wmma_bf16(af0, bf, acc[0][j]);
            acc[1][j] = wmma_bf16(af1, bf, acc[1][j]);
        }
        __syncthreads();
    }

    int lane = threadIdx.x & 31;
    int lr = lane & 15, hi = lane >> 4;
#pragma unroll
    for (int i = 0; i < 2; ++i) {
#pragma unroll
        for (int j = 0; j < 4; ++j) {
            int row0 = rowBlock + r0 + i * 16;
            int col0 = colBlock + c0 + j * 16;
#pragma unroll
            for (int r = 0; r < 8; ++r) {
                int m = row0 + r + 8 * hi;
                int n = col0 + lr;
                Zx[(size_t)m * N4H + n] = acc[i][j][r] + bias[n];
            }
        }
    }
}

// ---------------- sequential LSTM recurrence (one workgroup per side) ----------------
// block = 1024 threads (32 waves). Per step: z = Zx[t] + h_bf @ WhT (WMMA), then gates.
__global__ void lstm_rec_kernel(const float* __restrict__ Zx_L, const float* __restrict__ Zx_R,
                                const unsigned short* __restrict__ WhT_L, const unsigned short* __restrict__ WhT_R,
                                const int* __restrict__ len_L, const int* __restrict__ len_R,
                                float* h_L, float* h_R,
                                unsigned short* hbf_L, unsigned short* hbf_R,
                                float* c_L, float* c_R,
                                float* z_L, float* z_R) {
    int side = blockIdx.x;
    const float* Zx = side ? Zx_R : Zx_L;
    const unsigned short* WhT = side ? WhT_R : WhT_L;
    const int* lens = side ? len_R : len_L;
    float* hf = side ? h_R : h_L;
    unsigned short* hb = side ? hbf_R : hbf_L;
    float* cs = side ? c_R : c_L;
    float* zb = side ? z_R : z_L;

    int tid = threadIdx.x;
    for (int i = tid; i < Bb * Hh; i += blockDim.x) { hf[i] = 0.f; cs[i] = 0.f; hb[i] = 0; }
    __threadfence();
    __syncthreads();

    int wid = tid >> 5;
    int colW = wid * 64;  // each wave owns 64 output columns (4 tiles) x all 128 rows

    for (int t = 0; t < Ss; ++t) {
        const float* ZxT = Zx + (size_t)t * Bb * N4H;
        // phase 1: z = Zx[t] + h @ Wh
        for (int ct = 0; ct < 4; ++ct) {
            int col0 = colW + ct * 16;
            v8f acc[8];
#pragma unroll
            for (int rt = 0; rt < 8; ++rt) acc[rt] = load_c(ZxT, rt * 16, col0, N4H);
            for (int kk = 0; kk < KH / 32; ++kk) {
                int k0 = kk * 32;
                v16bf bf = load_frag(WhT, col0, KH, k0);
#pragma unroll
                for (int rt = 0; rt < 8; ++rt) {
                    v16bf af = load_frag(hb, rt * 16, KH, k0);
                    acc[rt] = wmma_bf16(af, bf, acc[rt]);
                }
            }
#pragma unroll
            for (int rt = 0; rt < 8; ++rt) store_c(zb, rt * 16, col0, N4H, acc[rt]);
        }
        __threadfence();
        __syncthreads();
        // phase 2: gates + masked state update
        for (int i = tid; i < Bb * Hh; i += blockDim.x) {
            int b = i >> 9, n = i & (Hh - 1);
            if (t < lens[b]) {
                const float* zr = zb + (size_t)b * N4H;
                float zi = zr[n], zj = zr[Hh + n], zf = zr[2 * Hh + n], zo = zr[3 * Hh + n];
                float cn = sigm(zf + 1.0f) * cs[i] + sigm(zi) * tanhf(zj);
                float hn = sigm(zo) * tanhf(cn);
                cs[i] = cn;
                hf[i] = hn;
                hb[i] = f2bf(hn);
            }
        }
        __threadfence();
        __syncthreads();
    }
}

// ---------------- dense head: logits = [h_l, h_r] @ W_dense + b_dense ----------------
__global__ void dense_kernel(const float* __restrict__ hL, const float* __restrict__ hR,
                             const float* __restrict__ Wd, const float* __restrict__ bd,
                             float* __restrict__ out) {
    int idx = blockIdx.x * blockDim.x + threadIdx.x;
    if (idx >= Bb * Cc) return;
    int b = idx / Cc, c = idx % Cc;
    float s = bd[c];
    for (int k = 0; k < Hh; ++k) s += hL[b * Hh + k] * Wd[k * Cc + c];
    for (int k = 0; k < Hh; ++k) s += hR[b * Hh + k] * Wd[(Hh + k) * Cc + c];
    out[idx] = s;
}

extern "C" void kernel_launch(void* const* d_in, const int* in_sizes, int n_in,
                              void* d_out, int out_size, void* d_ws, size_t ws_size,
                              hipStream_t stream) {
    (void)in_sizes; (void)n_in; (void)out_size; (void)ws_size;
    const int*   target_ids = (const int*)d_in[0];
    const int*   target_len = (const int*)d_in[1];
    const int*   left_ids   = (const int*)d_in[2];
    const int*   left_len   = (const int*)d_in[3];
    const int*   right_ids  = (const int*)d_in[4];
    const int*   right_len  = (const int*)d_in[5];
    const float* emb        = (const float*)d_in[6];
    const float* W_left     = (const float*)d_in[7];
    const float* b_left     = (const float*)d_in[8];
    const float* W_right    = (const float*)d_in[9];
    const float* b_right    = (const float*)d_in[10];
    const float* W_dense    = (const float*)d_in[11];
    const float* b_dense    = (const float*)d_in[12];
    float* out = (float*)d_out;

    char* ws = (char*)d_ws;
    size_t off = 0;
    auto alloc = [&](size_t bytes) -> void* {
        void* p = ws + off;
        off = (off + bytes + 255) & ~(size_t)255;
        return p;
    };

    float*          mean_t = (float*)alloc((size_t)Bb * Ee * 4);
    unsigned short* WxT_L  = (unsigned short*)alloc((size_t)N4H * KPAD * 2);
    unsigned short* WxT_R  = (unsigned short*)alloc((size_t)N4H * KPAD * 2);
    unsigned short* WhT_L  = (unsigned short*)alloc((size_t)N4H * KH * 2);
    unsigned short* WhT_R  = (unsigned short*)alloc((size_t)N4H * KH * 2);
    unsigned short* X_L    = (unsigned short*)alloc((size_t)MROWS * KPAD * 2);
    unsigned short* X_R    = (unsigned short*)alloc((size_t)MROWS * KPAD * 2);
    float*          Zx_L   = (float*)alloc((size_t)MROWS * N4H * 4);
    float*          Zx_R   = (float*)alloc((size_t)MROWS * N4H * 4);
    float*          h_L    = (float*)alloc((size_t)Bb * Hh * 4);
    float*          h_R    = (float*)alloc((size_t)Bb * Hh * 4);
    unsigned short* hbf_L  = (unsigned short*)alloc((size_t)Bb * Hh * 2);
    unsigned short* hbf_R  = (unsigned short*)alloc((size_t)Bb * Hh * 2);
    float*          c_L    = (float*)alloc((size_t)Bb * Hh * 4);
    float*          c_R    = (float*)alloc((size_t)Bb * Hh * 4);
    float*          z_L    = (float*)alloc((size_t)Bb * N4H * 4);
    float*          z_R    = (float*)alloc((size_t)Bb * N4H * 4);

    mean_target_kernel<<<Bb, 64, 0, stream>>>(target_ids, target_len, emb, mean_t);

    transpose_wx_kernel<<<2048, 256, 0, stream>>>(W_left, WxT_L);
    transpose_wx_kernel<<<2048, 256, 0, stream>>>(W_right, WxT_R);
    transpose_wh_kernel<<<2048, 256, 0, stream>>>(W_left, WhT_L);
    transpose_wh_kernel<<<2048, 256, 0, stream>>>(W_right, WhT_R);

    build_x_kernel<<<MROWS, 128, 0, stream>>>(left_ids, emb, mean_t, X_L);
    build_x_kernel<<<MROWS, 128, 0, stream>>>(right_ids, emb, mean_t, X_R);

    dim3 ggrid(MROWS / 64, N4H / 256);
    gemm_zx_kernel<<<ggrid, 256, 0, stream>>>(X_L, WxT_L, b_left, Zx_L);
    gemm_zx_kernel<<<ggrid, 256, 0, stream>>>(X_R, WxT_R, b_right, Zx_R);

    lstm_rec_kernel<<<2, 1024, 0, stream>>>(Zx_L, Zx_R, WhT_L, WhT_R, left_len, right_len,
                                            h_L, h_R, hbf_L, hbf_R, c_L, c_R, z_L, z_R);

    dense_kernel<<<(Bb * Cc + 127) / 128, 128, 0, stream>>>(h_L, h_R, W_dense, b_dense, out);
}
